// VisionMambaEncoder_51281909514440
// MI455X (gfx1250) — compile-verified, matrix operand order checked by reading, not run on hardware
//
#include <hip/hip_runtime.h>
#include <hip/hip_bf16.h>
#include <cstdint>

// ---------------- problem constants ----------------
#define BB 4
#define CIN 3
#define TT 8
#define IMG 224
#define PP 16
#define HP 14          // IMG/P
#define NPAT 196       // HP*HP
#define LSEQ 1568      // TT*NPAT
#define EE 384
#define DEPTH 24
#define DI 768         // D_INNER
#define NST 16
#define DTR 24
#define KC 4
#define MROWS (BB*LSEQ)   // 6272
#define FEATS_N ((size_t)MROWS*EE)

typedef __attribute__((ext_vector_type(16))) __bf16 v16bf;
typedef __attribute__((ext_vector_type(8)))  float  v8f;

struct U4 { unsigned int a,b,c,d; };
union Frag { v16bf v; U4 q[2]; };

__device__ __forceinline__ unsigned short f2bf(float f) {
    unsigned int u = __float_as_uint(f);
    unsigned int r = (u + 0x7FFFu + ((u >> 16) & 1u)) >> 16;  // RNE
    return (unsigned short)r;
}
__device__ __forceinline__ float sigm(float x){ return 1.f/(1.f+__expf(-x)); }

// ---------------- small utility kernels ----------------
__global__ __launch_bounds__(256) void k_fill0(float* p, size_t n){
    size_t i = (size_t)blockIdx.x*256 + threadIdx.x;
    size_t stride = (size_t)gridDim.x*256;
    for(; i<n; i+=stride) p[i]=0.f;
}

__global__ __launch_bounds__(256) void k_cvt_bf(const float* __restrict__ s,
                                                unsigned short* __restrict__ d, size_t n){
    size_t i=(size_t)blockIdx.x*256+threadIdx.x;
    size_t stride=(size_t)gridDim.x*256;
    for(; i<n; i+=stride) d[i]=f2bf(s[i]);
}

// convert + pad stacked weight tensors: src[Lc][Nsrc][Ksrc] -> dst[Lc][Npad][Kpad] (zero padded)
__global__ __launch_bounds__(256) void k_cvt_pad(const float* __restrict__ src,
        unsigned short* __restrict__ dst, int Lc,int Nsrc,int Ksrc,int Npad,int Kpad){
    size_t total=(size_t)Lc*Npad*Kpad;
    size_t i=(size_t)blockIdx.x*256+threadIdx.x;
    if(i>=total) return;
    int k=(int)(i%Kpad); size_t t=i/Kpad; int n=(int)(t%Npad); int l=(int)(t/Npad);
    float v=(n<Nsrc&&k<Ksrc)? src[((size_t)l*Nsrc+n)*Ksrc+k] : 0.f;
    dst[i]=f2bf(v);
}

// im2col: build bf16 A[MROWS][768]; K index = cin*256 + py*16 + px
__global__ __launch_bounds__(256) void k_im2col(const float* __restrict__ x,
                                                unsigned short* __restrict__ dst){
    size_t i=(size_t)blockIdx.x*256+threadIdx.x;
    size_t total=(size_t)MROWS*768;
    if(i>=total) return;
    int k=(int)(i%768); int m=(int)(i/768);
    int cin=k>>8; int r=k&255; int py=r>>4; int px=r&15;
    int b=m/(TT*NPAT); int rr=m%(TT*NPAT); int t=rr/NPAT; int n=rr%NPAT;
    int hp=n/HP; int wp=n%HP;
    size_t xi=((((size_t)b*CIN+cin)*TT+t)*IMG + (hp*PP+py))*IMG + (wp*PP+px);
    dst[i]=f2bf(x[xi]);
}

// ---------------- WMMA GEMM: C(MxN)=A(MxK)*W(NxK)^T, bf16 in, f32 out ----------
// block = 256 thr = 8 waves; wave tile 64(M)x32(N); block tile 128x128
#define EPI_NONE 0
#define EPI_PATCH 1
#define EPI_RELU 2
#define EPI_SOFTPLUS 3

template<int EPI>
__global__ __launch_bounds__(256) void k_gemm(
        const unsigned short* __restrict__ A, const unsigned short* __restrict__ W,
        float* __restrict__ C, int K, int Nstore, int ldc,
        const float* __restrict__ bias, const float* __restrict__ pos,
        const float* __restrict__ tpos){
    const int lane = threadIdx.x & 31;
    const int wid  = threadIdx.x >> 5;
    const int m0 = blockIdx.y*128 + (wid & 1)*64;
    const int n0 = blockIdx.x*128 + (wid >> 1)*32;
    const int rlane = lane & 15;
    const int ksel  = (lane >> 4) * 8;    // lanes 0-15: K 0..7 / 16..23 ; lanes 16-31: K 8..15 / 24..31

    v8f acc[4][2];
#pragma unroll
    for(int i=0;i<4;i++)
#pragma unroll
        for(int j=0;j<2;j++) acc[i][j]=(v8f){0.f,0.f,0.f,0.f,0.f,0.f,0.f,0.f};

    for(int k0=0;k0<K;k0+=32){
        Frag a[4], b[2];
#pragma unroll
        for(int i=0;i<4;i++){
            const unsigned short* p=A+(size_t)(m0+i*16+rlane)*K+k0+ksel;
            a[i].q[0]=*(const U4*)p;
            a[i].q[1]=*(const U4*)(p+16);
        }
#pragma unroll
        for(int j=0;j<2;j++){
            const unsigned short* p=W+(size_t)(n0+j*16+rlane)*K+k0+ksel;
            b[j].q[0]=*(const U4*)p;
            b[j].q[1]=*(const U4*)(p+16);
        }
#pragma unroll
        for(int i=0;i<4;i++)
#pragma unroll
            for(int j=0;j<2;j++)
                acc[i][j]=__builtin_amdgcn_wmma_f32_16x16x32_bf16(
                    false, a[i].v, false, b[j].v, (short)0, acc[i][j], false, false);
    }

    const int mo=(lane>>4)*8;
#pragma unroll
    for(int j=0;j<2;j++){
        int n=n0+j*16+rlane;
        if(n>=Nstore) continue;
#pragma unroll
        for(int i=0;i<4;i++){
#pragma unroll
            for(int r=0;r<8;r++){
                int m=m0+i*16+mo+r;
                float v=acc[i][j][r];
                if(EPI==EPI_PATCH){
                    int t=(m/NPAT)&7; int pidx=m%NPAT;
                    v += bias[n]+pos[(size_t)pidx*EE+n]+tpos[(size_t)t*EE+n];
                }else if(EPI==EPI_RELU){
                    v += bias[n]; v = v>0.f? v:0.f;
                }else if(EPI==EPI_SOFTPLUS){
                    v += bias[n]; v = v>20.f? v : log1pf(__expf(v));
                }
                C[(size_t)m*ldc+n]=v;
            }
        }
    }
}

// ---------------- per-row wave kernels (E=384 = 12 per lane) ----------------
__global__ __launch_bounds__(256) void k_addnorm(float* __restrict__ hidden,
        float* __restrict__ residual, const float* __restrict__ nw,
        unsigned short* __restrict__ hbf){
    int lane=threadIdx.x&31; int row=blockIdx.x*8+(threadIdx.x>>5);
    size_t base=(size_t)row*EE;
    float v[12]; float ss=0.f;
#pragma unroll
    for(int i=0;i<12;i++){ int e=i*32+lane;
        float x=hidden[base+e]+residual[base+e]; residual[base+e]=x; v[i]=x; ss+=x*x; }
    for(int o=16;o;o>>=1) ss += __shfl_xor(ss,o);
    float r=rsqrtf(ss*(1.f/EE)+1e-5f);
#pragma unroll
    for(int i=0;i<12;i++){ int e=i*32+lane; hbf[base+e]=f2bf(v[i]*r*nw[e]); }
}

__global__ __launch_bounds__(256) void k_final(const float* __restrict__ hidden,
        const float* __restrict__ residual, const float* __restrict__ nw,
        float* __restrict__ out){
    int lane=threadIdx.x&31; int row=blockIdx.x*8+(threadIdx.x>>5);
    size_t base=(size_t)row*EE;
    float v[12]; float ss=0.f;
#pragma unroll
    for(int i=0;i<12;i++){ int e=i*32+lane;
        float x=hidden[base+e]+residual[base+e]; v[i]=x; ss+=x*x; }
    for(int o=16;o;o>>=1) ss += __shfl_xor(ss,o);
    float r=rsqrtf(ss*(1.f/EE)+1e-5f);
#pragma unroll
    for(int i=0;i<12;i++){ int e=i*32+lane; out[base+e]=v[i]*r*nw[e]; }
}

// logits + sigmoid gating of hidden (xseq)
__global__ __launch_bounds__(256) void k_gate(const float* __restrict__ h1,
        const float* __restrict__ w2, const float* __restrict__ b2,
        float* __restrict__ hidden, float* __restrict__ outlog){
    int lane=threadIdx.x&31; int row=blockIdx.x*8+(threadIdx.x>>5);
    size_t base=(size_t)row*EE;
    float s=0.f;
#pragma unroll
    for(int i=0;i<12;i++){ int e=i*32+lane; s += h1[base+e]*w2[e]; }
    for(int o=16;o;o>>=1) s += __shfl_xor(s,o);
    float logit=s+b2[0];
    if(lane==0) outlog[row]=logit;
    float g=sigm(logit);
#pragma unroll
    for(int i=0;i<12;i++){ int e=i*32+lane; hidden[base+e]*=g; }
}

// ---------------- causal depthwise conv (K=4) + silu; dir=1 reads flipped --------
__global__ __launch_bounds__(256) void k_conv(const float* __restrict__ xz,
        const float* __restrict__ cw, const float* __restrict__ cb,
        float* __restrict__ xt, unsigned short* __restrict__ xtbf, int dir){
    size_t i=(size_t)blockIdx.x*256+threadIdx.x;
    size_t total=(size_t)MROWS*DI;
    if(i>=total) return;
    int d=(int)(i%DI); int row=(int)(i/DI);
    int b=row/LSEQ; int l=row%LSEQ;
    float acc=cb[d];
#pragma unroll
    for(int k=0;k<KC;k++){
        int lp=l-(KC-1)+k;
        if(lp>=0){
            int lsrc = dir ? (LSEQ-1-lp) : lp;
            acc += xz[((size_t)(b*LSEQ+lsrc))*(2*DI)+d]*cw[d*KC+k];
        }
    }
    float s=acc*sigm(acc);
    xt[i]=s; xtbf[i]=f2bf(s);
}

// dbc[:, :24] -> bf16 padded to 32 cols
__global__ __launch_bounds__(256) void k_dtprep(const float* __restrict__ dbc,
        unsigned short* __restrict__ dst){
    size_t i=(size_t)blockIdx.x*256+threadIdx.x;
    size_t total=(size_t)MROWS*32;
    if(i>=total) return;
    int j=(int)(i%32); int row=(int)(i/32);
    dst[i] = (j<DTR) ? f2bf(dbc[(size_t)row*64+j]) : (unsigned short)0;
}

// ---------------- selective scan: one thread per (b,d); LDS stages B/C -----------
__global__ __launch_bounds__(256) void k_scan(const float* __restrict__ xt,
        const float* __restrict__ dtm, const float* __restrict__ dbc,
        const float* __restrict__ Alog, const float* __restrict__ Dp,
        const float* __restrict__ xz, float* __restrict__ ybuf, int dir){
    int b = blockIdx.x/3;
    int d = (blockIdx.x%3)*256 + threadIdx.x;
    float Av[NST];
#pragma unroll
    for(int n=0;n<NST;n++) Av[n]=-__expf(Alog[(size_t)d*NST+n]);
    float h[NST];
#pragma unroll
    for(int n=0;n<NST;n++) h[n]=0.f;
    float Dv=Dp[d];
    __shared__ float sBC[32];
    for(int l=0;l<LSEQ;l++){
        size_t row=(size_t)(b*LSEQ+l);
        if(threadIdx.x<32) sBC[threadIdx.x]=dbc[row*64+24+threadIdx.x];
        __syncthreads();
        float dt=dtm[row*DI+d];
        float x =xt [row*DI+d];
        float dx=dt*x;
        float y=0.f;
#pragma unroll
        for(int n=0;n<NST;n++){
            h[n]=h[n]*__expf(dt*Av[n]) + dx*sBC[n];
            y += h[n]*sBC[NST+n];
        }
        y += x*Dv;
        int lorig = dir ? (LSEQ-1-l) : l;
        float zv = xz[((size_t)(b*LSEQ+lorig))*(2*DI)+DI+d];
        y *= zv*sigm(zv);
        size_t orow=(size_t)(b*LSEQ+lorig);
        if(dir) ybuf[orow*DI+d]+=y; else ybuf[orow*DI+d]=y;
        __syncthreads();
    }
}

// ---------------- host launch ----------------
extern "C" void kernel_launch(void* const* d_in, const int* in_sizes, int n_in,
                              void* d_out, int out_size, void* d_ws, size_t ws_size,
                              hipStream_t stream){
    const float* x          =(const float*)d_in[0];
    const float* patch_w    =(const float*)d_in[1];
    const float* patch_b    =(const float*)d_in[2];
    const float* pos_embed  =(const float*)d_in[3];
    const float* temporal   =(const float*)d_in[4];
    const float* imp_w1     =(const float*)d_in[5];
    const float* imp_b1     =(const float*)d_in[6];
    const float* imp_w2     =(const float*)d_in[7];
    const float* imp_b2     =(const float*)d_in[8];
    const float* in_w       =(const float*)d_in[9];
    const float* out_w      =(const float*)d_in[10];
    const float* norm_w     =(const float*)d_in[11];
    const float* norm_f_w   =(const float*)d_in[12];
    const float* conv_w[2]  ={(const float*)d_in[13],(const float*)d_in[20]};
    const float* conv_b[2]  ={(const float*)d_in[14],(const float*)d_in[21]};
    const float* xp_w[2]    ={(const float*)d_in[15],(const float*)d_in[22]};
    const float* dt_w[2]    ={(const float*)d_in[16],(const float*)d_in[23]};
    const float* dt_b[2]    ={(const float*)d_in[17],(const float*)d_in[24]};
    const float* A_log[2]   ={(const float*)d_in[18],(const float*)d_in[25]};
    const float* Dfac[2]    ={(const float*)d_in[19],(const float*)d_in[26]};

    float* out=(float*)d_out;

    char* ws=(char*)d_ws;
    size_t off=0;
    auto alloc=[&](size_t bytes)->char*{ char* p=ws+off; off+=(bytes+255)&~(size_t)255; return p; };

    unsigned short* im2col =(unsigned short*)alloc((size_t)MROWS*768*2);
    unsigned short* w_patch=(unsigned short*)alloc((size_t)EE*768*2);
    unsigned short* w_imp1 =(unsigned short*)alloc((size_t)EE*EE*2);
    unsigned short* w_in   =(unsigned short*)alloc((size_t)DEPTH*1536*EE*2);
    unsigned short* w_out  =(unsigned short*)alloc((size_t)DEPTH*EE*DI*2);
    unsigned short* w_xp[2], *w_dt[2];
    for(int d=0;d<2;d++){ w_xp[d]=(unsigned short*)alloc((size_t)DEPTH*128*DI*2);
                          w_dt[d]=(unsigned short*)alloc((size_t)DEPTH*DI*32*2); }
    float* hidden  =(float*)alloc((size_t)MROWS*EE*4);
    float* residual=(float*)alloc((size_t)MROWS*EE*4);
    unsigned short* hn_bf=(unsigned short*)alloc((size_t)MROWS*EE*2);
    float* xzb   =(float*)alloc((size_t)MROWS*1536*4);
    float* ybuf  =(float*)alloc((size_t)MROWS*DI*4);
    float* xt[2]; unsigned short* xtbf[2]; float* dbc[2]; unsigned short* dtlow[2]; float* dtb[2];
    for(int d=0;d<2;d++){
        xt[d]  =(float*)alloc((size_t)MROWS*DI*4);
        xtbf[d]=(unsigned short*)alloc((size_t)MROWS*DI*2);
        dbc[d] =(float*)alloc((size_t)MROWS*64*4);
        dtlow[d]=(unsigned short*)alloc((size_t)MROWS*32*2);
        dtb[d] =(float*)alloc((size_t)MROWS*DI*4);
    }
    float* h1 = ybuf;                 // reuse (h1 only used before layers)
    unsigned short* y_bf = im2col;    // reuse (im2col only used for patch embed)

    const int GM = MROWS/128;         // 49
    dim3 blk(256);

    // ---- weight conversion ----
    k_cvt_pad<<<((size_t)EE*768+255)/256,blk,0,stream>>>(patch_w,w_patch,1,EE,768,EE,768);
    k_cvt_pad<<<((size_t)EE*EE+255)/256,blk,0,stream>>>(imp_w1,w_imp1,1,EE,EE,EE,EE);
    k_cvt_pad<<<((size_t)DEPTH*1536*EE+255)/256,blk,0,stream>>>(in_w,w_in,1,DEPTH*1536,EE,DEPTH*1536,EE);
    k_cvt_pad<<<((size_t)DEPTH*EE*DI+255)/256,blk,0,stream>>>(out_w,w_out,1,DEPTH*EE,DI,DEPTH*EE,DI);
    for(int d=0;d<2;d++){
        k_cvt_pad<<<((size_t)DEPTH*128*DI+255)/256,blk,0,stream>>>(xp_w[d],w_xp[d],DEPTH,56,DI,128,DI);
        k_cvt_pad<<<((size_t)DEPTH*DI*32+255)/256,blk,0,stream>>>(dt_w[d],w_dt[d],DEPTH,DI,DTR,DI,32);
    }

    // ---- patch embed (+bias+pos+temporal) -> hidden ----
    k_im2col<<<((size_t)MROWS*768+255)/256,blk,0,stream>>>(x,im2col);
    k_gemm<EPI_PATCH><<<dim3(EE/128,GM),blk,0,stream>>>(im2col,w_patch,hidden,768,EE,EE,
                                                        patch_b,pos_embed,temporal);
    // ---- importance MLP + gating ----
    k_cvt_bf<<<18816,blk,0,stream>>>(hidden,hn_bf,(size_t)MROWS*EE);
    k_gemm<EPI_RELU><<<dim3(EE/128,GM),blk,0,stream>>>(hn_bf,w_imp1,h1,EE,EE,EE,
                                                       imp_b1,nullptr,nullptr);
    k_gate<<<MROWS/8,blk,0,stream>>>(h1,imp_w2,imp_b2,hidden,out+FEATS_N);

    // ---- residual = 0 ----
    k_fill0<<<4096,blk,0,stream>>>(residual,(size_t)MROWS*EE);

    // ---- 24 mamba layers ----
    for(int layer=0;layer<DEPTH;layer++){
        k_addnorm<<<MROWS/8,blk,0,stream>>>(hidden,residual,norm_w+(size_t)layer*EE,hn_bf);
        k_gemm<EPI_NONE><<<dim3(1536/128,GM),blk,0,stream>>>(hn_bf,w_in+(size_t)layer*1536*EE,
                xzb,EE,1536,1536,nullptr,nullptr,nullptr);
        for(int dir=0;dir<2;dir++){
            k_conv<<<((size_t)MROWS*DI+255)/256,blk,0,stream>>>(xzb,
                    conv_w[dir]+(size_t)layer*DI*KC, conv_b[dir]+(size_t)layer*DI,
                    xt[dir],xtbf[dir],dir);
            k_gemm<EPI_NONE><<<dim3(1,GM),blk,0,stream>>>(xtbf[dir],
                    w_xp[dir]+(size_t)layer*128*DI, dbc[dir],DI,64,64,
                    nullptr,nullptr,nullptr);
            k_dtprep<<<((size_t)MROWS*32+255)/256,blk,0,stream>>>(dbc[dir],dtlow[dir]);
            k_gemm<EPI_SOFTPLUS><<<dim3(DI/128,GM),blk,0,stream>>>(dtlow[dir],
                    w_dt[dir]+(size_t)layer*DI*32, dtb[dir],32,DI,DI,
                    dt_b[dir]+(size_t)layer*DI,nullptr,nullptr);
            k_scan<<<BB*3,blk,0,stream>>>(xt[dir],dtb[dir],dbc[dir],
                    A_log[dir]+(size_t)layer*DI*NST, Dfac[dir]+(size_t)layer*DI,
                    xzb,ybuf,dir);
        }
        k_cvt_bf<<<18816,blk,0,stream>>>(ybuf,y_bf,(size_t)MROWS*DI);
        k_gemm<EPI_NONE><<<dim3(EE/128,GM),blk,0,stream>>>(y_bf,w_out+(size_t)layer*EE*DI,
                hidden,DI,EE,EE,nullptr,nullptr,nullptr);
    }

    // ---- final rmsnorm -> feats ----
    k_final<<<MROWS/8,blk,0,stream>>>(hidden,residual,norm_f_w,out);
}
